// DynamicEdgeConvPN_18485539242026
// MI455X (gfx1250) — matrix operand order repaired and checked by reading, not compiled, not run on hardware
//
#include <hip/hip_runtime.h>

#define BB   32
#define NN   2048
#define KNBR 16
#define FIN  32
#define FOUT 64
#define NPTS (BB * NN)

typedef _Float16 v16h __attribute__((ext_vector_type(16)));
typedef _Float16 v8h  __attribute__((ext_vector_type(8)));
typedef _Float16 h2   __attribute__((ext_vector_type(2)));
typedef float    v8f  __attribute__((ext_vector_type(8)));

// ---------------------------------------------------------------------------
// Kernel 0: one-time f32 -> packed-f16 conversion of x (B*N x 32).
// Lets the WMMA kernels load A-tiles as raw b128 with zero per-point cvt.
// ---------------------------------------------------------------------------
__global__ __launch_bounds__(256) void cvt_kernel(const float* __restrict__ x,
                                                  _Float16* __restrict__ xh)
{
    const int i = blockIdx.x * blockDim.x + threadIdx.x;   // pair index
    const float2 f = ((const float2*)x)[i];
    h2 p;
    p[0] = (_Float16)f.x;
    p[1] = (_Float16)f.y;
    ((h2*)xh)[i] = p;
}

// ---------------------------------------------------------------------------
// Kernel 1: brute-force kNN (K=16) per batch of 2048 points in 3-D.
// One thread per query; positions tiled through LDS; sorted top-16 kept in
// registers via fully-unrolled compare-swap insertion. Stores GLOBAL rows.
// (max-over-K downstream is permutation-invariant, so any top-16 set matches.)
// ---------------------------------------------------------------------------
__global__ __launch_bounds__(256) void knn_kernel(const float* __restrict__ pos,
                                                  int* __restrict__ knn)
{
    __shared__ float sp[256 * 4];               // x,y,z,|p|^2 per tile point

    const int q    = blockIdx.x * 256 + threadIdx.x;   // global query row
    const int b    = q / NN;                           // 256 | 2048 -> uniform
    const int base = b * NN;

    const float qx = pos[q * 3 + 0];
    const float qy = pos[q * 3 + 1];
    const float qz = pos[q * 3 + 2];
    const float qs = qx * qx + qy * qy + qz * qz;

    float bd[KNBR];
    int   bi[KNBR];
#pragma unroll
    for (int j = 0; j < KNBR; ++j) { bd[j] = 3e38f; bi[j] = base; }

    for (int t = 0; t < NN; t += 256) {
        const int row = base + t + threadIdx.x;
        const float px = pos[row * 3 + 0];
        const float py = pos[row * 3 + 1];
        const float pz = pos[row * 3 + 2];
        sp[threadIdx.x * 4 + 0] = px;
        sp[threadIdx.x * 4 + 1] = py;
        sp[threadIdx.x * 4 + 2] = pz;
        sp[threadIdx.x * 4 + 3] = px * px + py * py + pz * pz;
        __syncthreads();

        for (int m = 0; m < 256; ++m) {
            const int mrow = base + t + m;
            float d = qs + sp[m * 4 + 3]
                    - 2.0f * (qx * sp[m * 4 + 0] + qy * sp[m * 4 + 1] + qz * sp[m * 4 + 2]);
            if (mrow == q) d = 3e38f;              // exclude self (eye*1e30)
            if (d < bd[KNBR - 1]) {
                float cd = d; int ci = mrow;
#pragma unroll
                for (int j = 0; j < KNBR; ++j) {   // sorted insert, reg-only
                    if (cd < bd[j]) {
                        float td = bd[j]; bd[j] = cd; cd = td;
                        int   ti = bi[j]; bi[j] = ci; ci = ti;
                    }
                }
            }
        }
        __syncthreads();
    }
#pragma unroll
    for (int j = 0; j < KNBR; ++j) knn[q * KNBR + j] = bi[j];
}

// A-tile helper: for the 16-bit 16x32 A layout, half h of the wave holds
// contiguous K [16h..16h+15] of its row (matching the B fill) -> two
// aligned 16-byte loads, no shuffling, no conversion.
__device__ __forceinline__ v16h load_arow(const _Float16* __restrict__ row, int h)
{
    const v8h lo = *(const v8h*)(row + 16 * h);
    const v8h hi = *(const v8h*)(row + 16 * h + 8);
    v16h a;
#pragma unroll
    for (int e = 0; e < 8; ++e) { a[e] = lo[e]; a[8 + e] = hi[e]; }
    return a;
}

// ---------------------------------------------------------------------------
// Kernel 2: fused gather + edge-MLP (WMMA) + relu + max-over-K.
// One wave per point per iteration: A0 = x_i broadcast (2x b128 loads),
// A1 = x_j - x_i (packed f16 sub); W_edge held in registers as 2x4 B-tiles;
// 8x v_wmma_f32_16x16x32_f16 per point. Bias lives in persistent accumulator
// registers passed as C to the first WMMA of each chain -> zero per-point
// accumulator-init movs. Writes out = edge_max.
// ---------------------------------------------------------------------------
__global__ __launch_bounds__(256) void edge_kernel(const _Float16* __restrict__ xh,
                                                   const float* __restrict__ W_edge,
                                                   const float* __restrict__ b_edge,
                                                   const int* __restrict__ knn,
                                                   float* __restrict__ out)
{
    const int lane = threadIdx.x & 31;
    const int h    = lane >> 4;          // half of wave (K-chunk selector)
    const int r    = lane & 15;          // row (neighbor) / col within tile
    const int wavesPerBlock = blockDim.x >> 5;
    const int wid    = blockIdx.x * wavesPerBlock + (threadIdx.x >> 5);
    const int nwaves = gridDim.x * wavesPerBlock;

    // ---- preload W_edge as B-tiles [k-half][n-tile], f16 --------------------
    // B as filled here (contiguous K, same order as A): lane n = lane&15 is column; lanes 0-15 hold K=0..15
    // (vgpr v -> K=2v,2v+1), lanes 16-31 hold K=16..31.
    v16h Bt[2][4];
#pragma unroll
    for (int kk = 0; kk < 2; ++kk) {
#pragma unroll
        for (int t = 0; t < 4; ++t) {
#pragma unroll
            for (int v = 0; v < 8; ++v) {
                const int kb  = kk * 32 + (h << 4) + 2 * v;
                const int col = t * 16 + r;
                Bt[kk][t][2 * v]     = (_Float16)W_edge[kb * FOUT + col];
                Bt[kk][t][2 * v + 1] = (_Float16)W_edge[(kb + 1) * FOUT + col];
            }
        }
    }

    // Persistent bias accumulators: C operand of the first WMMA of each tile.
    v8f biasC[4];
#pragma unroll
    for (int t = 0; t < 4; ++t) {
        const float be = b_edge[t * 16 + r];
#pragma unroll
        for (int j = 0; j < 8; ++j) biasC[t][j] = be;
    }

    const int c0 = h * 32 + r;           // this lane's output cols (tiles 2h, 2h+1)
    const int c1 = c0 + 16;

    for (int g = wid; g < NPTS; g += nwaves) {
        const v16h A0 = load_arow(xh + g * FIN, h);          // msg cols 0..31

        const int nb = knn[g * KNBR + r];                    // row r = neighbor r
        const v16h Aj = load_arow(xh + nb * FIN, h);
        const v16h A1 = Aj - A0;                             // msg cols 32..63

        v8f acc[4];
#pragma unroll
        for (int t = 0; t < 4; ++t)
            acc[t] = __builtin_amdgcn_wmma_f32_16x16x32_f16(
                false, A0, false, Bt[0][t], (short)0, biasC[t], false, false);
#pragma unroll
        for (int t = 0; t < 4; ++t)
            acc[t] = __builtin_amdgcn_wmma_f32_16x16x32_f16(
                false, A1, false, Bt[1][t], (short)0, acc[t], false, false);

        // max_k relu(h) == relu(max_k h): per-lane max of 8 D rows, then
        // combine the two halves (rows 0-7 vs 8-15) with a cross-half shuffle.
        float mx[4];
#pragma unroll
        for (int t = 0; t < 4; ++t) {
            float m = 0.0f;                          // relu floor
#pragma unroll
            for (int j = 0; j < 8; ++j) m = fmaxf(m, acc[t][j]);
            m = fmaxf(m, __shfl_xor(m, 16, 32));
            mx[t] = m;
        }

        out[g * FOUT + c0] = mx[2 * h];
        out[g * FOUT + c1] = mx[2 * h + 1];
    }
}

// ---------------------------------------------------------------------------
// Kernel 3: skip MLP via WMMA. One wave handles 16 points: A = 16 points x 32
// features, B = W_nn (32x64) as 4 register tiles -> 4 WMMAs per 16 points.
// Bias passed as the C operand (persistent registers). relu, then out += skip
// (read-modify-write; each (row,col) owned by exactly one lane, edge_kernel
// wrote out first in stream order).
// ---------------------------------------------------------------------------
__global__ __launch_bounds__(256) void skip_kernel(const _Float16* __restrict__ xh,
                                                   const float* __restrict__ W_nn,
                                                   const float* __restrict__ b_nn,
                                                   float* __restrict__ out)
{
    const int lane = threadIdx.x & 31;
    const int h    = lane >> 4;
    const int r    = lane & 15;
    const int wavesPerBlock = blockDim.x >> 5;
    const int wid    = blockIdx.x * wavesPerBlock + (threadIdx.x >> 5);
    const int nwaves = gridDim.x * wavesPerBlock;

    v16h Bn[4];
#pragma unroll
    for (int t = 0; t < 4; ++t) {
#pragma unroll
        for (int v = 0; v < 8; ++v) {
            const int kb  = (h << 4) + 2 * v;          // K = 0..31
            const int col = t * 16 + r;
            Bn[t][2 * v]     = (_Float16)W_nn[kb * FOUT + col];
            Bn[t][2 * v + 1] = (_Float16)W_nn[(kb + 1) * FOUT + col];
        }
    }

    v8f biasC[4];
#pragma unroll
    for (int t = 0; t < 4; ++t) {
        const float bn = b_nn[t * 16 + r];
#pragma unroll
        for (int j = 0; j < 8; ++j) biasC[t][j] = bn;
    }

    const int ngroups = NPTS / 16;                     // 4096 groups of 16 points
    for (int grp = wid; grp < ngroups; grp += nwaves) {
        const int g0 = grp * 16;

        // A row r = point g0+r (both halves load the same row, different K-chunk)
        const v16h A = load_arow(xh + (g0 + r) * FIN, h);

        v8f acc[4];
#pragma unroll
        for (int t = 0; t < 4; ++t)
            acc[t] = __builtin_amdgcn_wmma_f32_16x16x32_f16(
                false, A, false, Bn[t], (short)0, biasC[t], false, false);

        // relu + accumulate into out (D row j+8h of the group, col 16t+r)
#pragma unroll
        for (int t = 0; t < 4; ++t) {
#pragma unroll
            for (int j = 0; j < 8; ++j) {
                const int row = g0 + j + 8 * h;
                const int col = t * 16 + r;
                out[row * FOUT + col] += fmaxf(acc[t][j], 0.0f);
            }
        }
    }
}

// ---------------------------------------------------------------------------
extern "C" void kernel_launch(void* const* d_in, const int* in_sizes, int n_in,
                              void* d_out, int out_size, void* d_ws, size_t ws_size,
                              hipStream_t stream)
{
    const float* x      = (const float*)d_in[0];
    const float* pos    = (const float*)d_in[1];
    const float* W_edge = (const float*)d_in[2];
    const float* b_edge = (const float*)d_in[3];
    const float* W_nn   = (const float*)d_in[4];
    const float* b_nn   = (const float*)d_in[5];
    // d_in[6] = batch: implicit in the fixed B,N layout

    // ws layout: [0, 4MB) knn int32; [4MB, 8MB) x as packed f16
    int*      knn = (int*)d_ws;
    _Float16* xh  = (_Float16*)((char*)d_ws + (size_t)NPTS * KNBR * sizeof(int));

    cvt_kernel <<<(NPTS * FIN / 2) / 256, 256, 0, stream>>>(x, xh);
    knn_kernel <<<NPTS / 256,            256, 0, stream>>>(pos, knn);
    edge_kernel<<<512,                   256, 0, stream>>>(xh, W_edge, b_edge, knn,
                                                           (float*)d_out);
    skip_kernel<<<512,                   256, 0, stream>>>(xh, W_nn, b_nn,
                                                           (float*)d_out);
}
